// GFASTKAN_Nodes_51187420234502
// MI455X (gfx1250) — matrix operand order repaired and checked by reading, not compiled, not run on hardware
//
#include <hip/hip_runtime.h>
#include <hip/hip_bf16.h>

// Problem constants (from reference)
#define NN 50000
#define DD 128
#define HH 128
#define CC 40
#define LL 3
#define GG 4
#define EE 800000
#define XCATW (DD + LL*HH)          // 512
#define KLAYER (DD*GG + DD)         // 640
#define KOUT   (XCATW*GG + XCATW)   // 2560
#define COUTPAD 48                  // 40 padded to 3 WMMA n-tiles

typedef __attribute__((ext_vector_type(16))) __bf16 v16bf;
typedef __attribute__((ext_vector_type(8)))  __bf16 v8bf;
typedef __attribute__((ext_vector_type(8)))  float  v8f;

__device__ __forceinline__ float wave_sum32(float v) {
#pragma unroll
    for (int o = 16; o > 0; o >>= 1) v += __shfl_xor(v, o, 32);
    return v;
}

// ---------------------------------------------------------------------------
// Fused LayerNorm + Gaussian RBF basis + SiLU base: one wave32 per row.
// Writes bf16 row [basis(Din*4) | silu(Din)] of length Din*5 into AB.
// ---------------------------------------------------------------------------
__global__ void fkan_basis_kernel(const float* __restrict__ xin, int ldin, int Din,
                                  const float* __restrict__ lng,
                                  const float* __restrict__ lnb,
                                  __bf16* __restrict__ AB, int ldAB) {
    int wid  = (blockIdx.x * blockDim.x + threadIdx.x) >> 5;
    int lane = threadIdx.x & 31;
    if (wid >= NN) return;
    const float* xr = xin + (size_t)wid * ldin;
    int chunk = Din >> 5;                       // 4 (layers) or 16 (output)
    float s = 0.f, ss = 0.f;
    for (int i = 0; i < chunk; ++i) {
        float v = xr[lane * chunk + i];
        s += v; ss += v * v;
    }
    s  = wave_sum32(s);
    ss = wave_sum32(ss);
    float mean = s / (float)Din;
    float var  = ss / (float)Din - mean * mean; // biased var, like jnp.var
    float rstd = rsqrtf(var + 1e-5f);

    __bf16* out = AB + (size_t)wid * ldAB;
    const float grid0 = -2.f, gstep = 4.f / 3.f;   // linspace(-2,2,4)
    const float invden = 0.75f;                    // 1 / (4/(G-1))
    for (int i = 0; i < chunk; ++i) {
        int d = lane * chunk + i;
        float xv = xr[d];
        float xn = (xv - mean) * rstd * lng[d] + lnb[d];
#pragma unroll
        for (int g = 0; g < GG; ++g) {
            float t = (xn - (grid0 + (float)g * gstep)) * invden;
            out[d * GG + g] = (__bf16)__expf(-t * t);
        }
        float sig = 1.f / (1.f + __expf(-xv));
        out[Din * GG + d] = (__bf16)(xv * sig);   // silu base input
    }
}

// ---------------------------------------------------------------------------
// WMMA bf16 GEMM + bias:  C[M,Nc] = A[M,K] * W[Nc,K]^T + bias
// One wave computes a 16 x (NT*16) strip: the A fragment is loaded once per
// K-step and reused for NT v_wmma_f32_16x16x32_bf16 issues (register-blocked
// N), doubling the wmma:vmem ratio vs one-tile-per-wave.
// A/B lane->element mapping: contiguous k = 16*half + s in both operands.
// ---------------------------------------------------------------------------
template <int NT>
__global__ void wmma_gemm_bias_kernel(const __bf16* __restrict__ A, int ldA, int K,
                                      const __bf16* __restrict__ W, int ldW,
                                      const float* __restrict__ bias, int nbias,
                                      float* __restrict__ Cout, int ldc,
                                      int ncols_store, int ngroups_n) {
    int wid  = (blockIdx.x * blockDim.x + threadIdx.x) >> 5;
    int lane = threadIdx.x & 31;
    int tg = wid % ngroups_n;                   // group of NT n-tiles
    int tm = wid / ngroups_n;
    if (tm >= NN / 16) return;                  // 50000 = 16*3125 exact
    int m0 = tm * 16, n0 = tg * NT * 16;
    int half = lane >> 4;                       // 0: lanes 0-15, 1: lanes 16-31
    int l15  = lane & 15;

    const __bf16* arow = A + (size_t)(m0 + l15) * ldA;   // A row M=m0+l15
    const __bf16* brow[NT];
#pragma unroll
    for (int t = 0; t < NT; ++t)                         // B col N=n0+t*16+l15
        brow[t] = W + (size_t)(n0 + t * 16 + l15) * ldW;

    v8f acc[NT];
#pragma unroll
    for (int t = 0; t < NT; ++t) acc[t] = (v8f){0.f,0.f,0.f,0.f,0.f,0.f,0.f,0.f};

    for (int k0 = 0; k0 < K; k0 += 32) {
        __builtin_prefetch(arow + k0 + 256, 0, 0);       // global_prefetch_b8
        // A 16x32 bf16: lane<16 -> K 0..15; lane>=16 -> K 16..31
        v8bf alo = *(const v8bf*)(arow + k0 + half * 16);
        v8bf ahi = *(const v8bf*)(arow + k0 + half * 16 + 8);
        v16bf a;
#pragma unroll
        for (int i = 0; i < 8; ++i) { a[i] = alo[i]; a[8 + i] = ahi[i]; }
#pragma unroll
        for (int t = 0; t < NT; ++t) {
            // B 32x16 bf16: lane<16 -> K 0..15 of col N; lane>=16 -> K 16..31
            v16bf b = *(const v16bf*)(brow[t] + k0 + half * 16);
            acc[t] = __builtin_amdgcn_wmma_f32_16x16x32_bf16(
                false, a, false, b, (short)0, acc[t], false, false);
        }
    }
    // C/D: VGPR v -> row m0 + v + 8*half, col n0 + t*16 + l15
#pragma unroll
    for (int t = 0; t < NT; ++t) {
#pragma unroll
        for (int v = 0; v < 8; ++v) {
            int row = m0 + half * 8 + v;
            int col = n0 + t * 16 + l15;
            if (col < ncols_store) {
                float bval = (col < nbias) ? bias[col] : 0.f;
                Cout[(size_t)row * ldc + col] = acc[t][v] + bval;
            }
        }
    }
}

// ---------------------------------------------------------------------------
// Weight packing: [sw | bw] -> bf16 row-major [Nout, K]
// ---------------------------------------------------------------------------
__global__ void pack_w_layer_kernel(const float* __restrict__ sw,
                                    const float* __restrict__ bw,
                                    __bf16* __restrict__ Wc) {
    int idx = blockIdx.x * blockDim.x + threadIdx.x;
    if (idx >= HH * KLAYER) return;
    int h = idx / KLAYER, j = idx % KLAYER;
    float v = (j < DD * GG) ? sw[h * (DD * GG) + j] : bw[h * DD + (j - DD * GG)];
    Wc[idx] = (__bf16)v;
}

__global__ void pack_w_out_kernel(const float* __restrict__ sw,
                                  const float* __restrict__ bw,
                                  __bf16* __restrict__ Wc) {
    int idx = blockIdx.x * blockDim.x + threadIdx.x;
    if (idx >= COUTPAD * KOUT) return;
    int r = idx / KOUT, j = idx % KOUT;
    float v = 0.f;                               // pad rows 40..47 with zeros
    if (r < CC)
        v = (j < XCATW * GG) ? sw[r * (XCATW * GG) + j]
                             : bw[r * XCATW + (j - XCATW * GG)];
    Wc[idx] = (__bf16)v;
}

// ---------------------------------------------------------------------------
// Elementwise / graph kernels
// ---------------------------------------------------------------------------
__global__ void copy_x_kernel(const float* __restrict__ x, float* __restrict__ xcat) {
    int i = blockIdx.x * blockDim.x + threadIdx.x;
    if (i >= NN * DD) return;
    int n = i / DD, d = i % DD;
    xcat[(size_t)n * XCATW + d] = x[i];
}

__global__ void deg_init_kernel(float* __restrict__ deg) {
    int i = blockIdx.x * blockDim.x + threadIdx.x;
    if (i < NN) deg[i] = 1.f;                    // self-loop contributes 1
}
__global__ void deg_acc_kernel(const int* __restrict__ dst, float* __restrict__ deg) {
    int e = blockIdx.x * blockDim.x + threadIdx.x;
    if (e < EE) atomicAdd(&deg[dst[e]], 1.f);    // global_atomic_add_f32
}
__global__ void deg_fin_kernel(float* __restrict__ deg) {
    int i = blockIdx.x * blockDim.x + threadIdx.x;
    if (i < NN) deg[i] = rsqrtf(deg[i]);         // deg >= 1 always
}

// self-loop term: out[n] = dis[n]^2 * h[n]
__global__ void prop_init_kernel(const float* __restrict__ hin,
                                 const float* __restrict__ dis,
                                 float* __restrict__ pout) {
    int i = blockIdx.x * blockDim.x + threadIdx.x;
    if (i >= NN * HH) return;
    int n = i / HH;
    float dn = dis[n];
    pout[i] = dn * dn * hin[i];
}

// one wave per edge: scatter-add dis[s]*dis[d]*h[s] into row d
__global__ void prop_edges_kernel(const int* __restrict__ src,
                                  const int* __restrict__ dst,
                                  const float* __restrict__ hin,
                                  const float* __restrict__ dis,
                                  float* __restrict__ pout) {
    int wid  = (blockIdx.x * blockDim.x + threadIdx.x) >> 5;
    int lane = threadIdx.x & 31;
    if (wid >= EE) return;
    int s = src[wid], d = dst[wid];
    float norm = dis[s] * dis[d];
    const float* hr = hin + (size_t)s * HH;
    float* pr = pout + (size_t)d * HH;
#pragma unroll
    for (int i = 0; i < HH / 32; ++i) {
        int c = lane + i * 32;
        atomicAdd(&pr[c], hr[c] * norm);
    }
}

__global__ void bn_zero_kernel(float* __restrict__ stats) {
    int i = threadIdx.x;
    if (i < 2 * HH) stats[i] = 0.f;
}
// 128 threads/block (one per column), each block reduces 128 rows
__global__ void bn_stats_kernel(const float* __restrict__ p, float* __restrict__ stats) {
    int c  = threadIdx.x;
    int r0 = blockIdx.x * 128;
    int r1 = r0 + 128; if (r1 > NN) r1 = NN;
    float s = 0.f, ss = 0.f;
    for (int r = r0; r < r1; ++r) {
        float v = p[(size_t)r * HH + c];
        s += v; ss += v * v;
    }
    atomicAdd(&stats[c], s);
    atomicAdd(&stats[HH + c], ss);
}
// NOTE: gcn_bias is a per-column constant added before BN; BN subtracts the
// column mean, so it cancels exactly and is intentionally omitted.
__global__ void bn_apply_kernel(const float* __restrict__ p,
                                const float* __restrict__ stats,
                                const float* __restrict__ g,
                                const float* __restrict__ b,
                                float* __restrict__ xcat, int colofs) {
    int i = blockIdx.x * blockDim.x + threadIdx.x;
    if (i >= NN * HH) return;
    int n = i / HH, c = i % HH;
    float mean = stats[c] / (float)NN;
    float var  = stats[HH + c] / (float)NN - mean * mean;
    float sc = g[c] * rsqrtf(var + 1e-5f);
    xcat[(size_t)n * XCATW + colofs + c] = (p[i] - mean) * sc + b[c];
}

// ---------------------------------------------------------------------------
extern "C" void kernel_launch(void* const* d_in, const int* in_sizes, int n_in,
                              void* d_out, int out_size, void* d_ws, size_t ws_size,
                              hipStream_t stream) {
    (void)in_sizes; (void)n_in; (void)out_size; (void)ws_size;
    const float* x        = (const float*)d_in[0];
    const int*   ei       = (const int*)d_in[1];   // [2, E]
    const float* ln_g     = (const float*)d_in[2];
    const float* ln_b     = (const float*)d_in[3];
    const float* sw       = (const float*)d_in[4];
    const float* bw       = (const float*)d_in[5];
    const float* bb       = (const float*)d_in[6];
    /* d_in[7] = gcn_bias: cancels through BatchNorm, unused */
    const float* bn_g     = (const float*)d_in[8];
    const float* bn_b     = (const float*)d_in[9];
    const float* ln_g_out = (const float*)d_in[10];
    const float* ln_b_out = (const float*)d_in[11];
    const float* sw_out   = (const float*)d_in[12];
    const float* bw_out   = (const float*)d_in[13];
    const float* bb_out   = (const float*)d_in[14];
    float* out = (float*)d_out;

    // Workspace carve-out (256B aligned slots)
    char* wp = (char*)d_ws;
    auto carve = [&](size_t bytes) -> void* {
        void* p = (void*)wp;
        wp += (bytes + 255) & ~(size_t)255;
        return p;
    };
    float*  xcat  = (float*)carve((size_t)NN * XCATW * 4);   // 102.4 MB
    __bf16* AB    = (__bf16*)carve((size_t)NN * KOUT * 2);   // 256 MB (reused)
    __bf16* Wc    = (__bf16*)carve((size_t)COUTPAD * KOUT * 2);
    float*  fk    = (float*)carve((size_t)NN * HH * 4);      // FastKAN out
    float*  pr    = (float*)carve((size_t)NN * HH * 4);      // GCN prop out
    float*  dis   = (float*)carve((size_t)NN * 4);           // rsqrt(deg)
    float*  stats = (float*)carve((size_t)2 * HH * 4);       // BN sum/sumsq

    const int* src = ei;
    const int* dst = ei + EE;

    copy_x_kernel<<<(NN * DD + 255) / 256, 256, 0, stream>>>(x, xcat);

    // Degrees (identical across layers: compute once)
    deg_init_kernel<<<(NN + 255) / 256, 256, 0, stream>>>(dis);
    deg_acc_kernel<<<(EE + 255) / 256, 256, 0, stream>>>(dst, dis);
    deg_fin_kernel<<<(NN + 255) / 256, 256, 0, stream>>>(dis);

    for (int i = 0; i < LL; ++i) {
        // Layer input is xcat cols [i*128, i*128+128): x for i=0, h_{i-1} after.
        fkan_basis_kernel<<<(NN + 7) / 8, 256, 0, stream>>>(
            xcat + i * DD, XCATW, DD, ln_g + i * DD, ln_b + i * DD, AB, KLAYER);
        pack_w_layer_kernel<<<(HH * KLAYER + 255) / 256, 256, 0, stream>>>(
            sw + (size_t)i * HH * DD * GG, bw + (size_t)i * HH * DD, Wc);
        // 16x64 strip per wave: 2 n-groups of 4 tiles -> 6250 waves
        int ngroups = HH / (4 * 16);
        int waves = (NN / 16) * ngroups;
        wmma_gemm_bias_kernel<4><<<(waves + 7) / 8, 256, 0, stream>>>(
            AB, KLAYER, KLAYER, Wc, KLAYER, bb + i * HH, HH, fk, HH, HH, ngroups);

        prop_init_kernel<<<(NN * HH + 255) / 256, 256, 0, stream>>>(fk, dis, pr);
        prop_edges_kernel<<<(EE + 7) / 8, 256, 0, stream>>>(src, dst, fk, dis, pr);

        bn_zero_kernel<<<1, 256, 0, stream>>>(stats);
        bn_stats_kernel<<<(NN + 127) / 128, 128, 0, stream>>>(pr, stats);
        bn_apply_kernel<<<(NN * HH + 255) / 256, 256, 0, stream>>>(
            pr, stats, bn_g + i * HH, bn_b + i * HH, xcat, DD + i * HH);
    }

    // Output FastKAN on the concat [N, 512] -> [N, 40]
    fkan_basis_kernel<<<(NN + 7) / 8, 256, 0, stream>>>(
        xcat, XCATW, XCATW, ln_g_out, ln_b_out, AB, KOUT);
    pack_w_out_kernel<<<(COUTPAD * KOUT + 255) / 256, 256, 0, stream>>>(
        sw_out, bw_out, Wc);
    // 16x48 strip per wave: all 3 n-tiles in one group -> the 256 MB basis
    // buffer streams from HBM exactly once.
    int waves = NN / 16;
    wmma_gemm_bias_kernel<3><<<(waves + 7) / 8, 256, 0, stream>>>(
        AB, KOUT, KOUT, Wc, KOUT, bb_out, CC, out, CC, CC, 1);
}
